// AttentionCritic_25434796326948
// MI455X (gfx1250) — matrix operand order repaired and checked by reading, not compiled, not run on hardware
//
#include <hip/hip_runtime.h>

#define N_AG   8
#define BATCH  32768
#define SDIM   128
#define ADIM   16
#define HID    128
#define HEADS  4
#define AD     32
#define IDIM   144
#define IPAD   160   // IDIM zero-padded to a multiple of 32 (WMMA K)

typedef __attribute__((ext_vector_type(16))) __bf16   v16bf;
typedef __attribute__((ext_vector_type(8)))  float    v8f;
typedef __attribute__((ext_vector_type(4)))  unsigned u32x4;

union Frag { v16bf v; u32x4 q[2]; };

// ---- helpers -------------------------------------------------------------

__device__ __forceinline__ unsigned short f2bf(float f) {
  unsigned u = __builtin_bit_cast(unsigned, f);
  u = (u + 0x7FFFu + ((u >> 16) & 1u)) >> 16;   // round-to-nearest-even
  return (unsigned short)u;
}
__device__ __forceinline__ float bflo(unsigned p) {
  return __builtin_bit_cast(float, p << 16);
}
__device__ __forceinline__ float bfhi(unsigned p) {
  return __builtin_bit_cast(float, p & 0xFFFF0000u);
}
__device__ __forceinline__ unsigned pk2(float a, float b) {
  return (unsigned)f2bf(a) | ((unsigned)f2bf(b) << 16);
}
__device__ __forceinline__ v8f vzero8() {
  v8f z;
#pragma unroll
  for (int r = 0; r < 8; ++r) z[r] = 0.f;
  return z;
}
__device__ __forceinline__ v8f wmma_bf16(const Frag& a, const Frag& b, v8f c) {
  return __builtin_amdgcn_wmma_f32_16x16x32_bf16(false, a.v, false, b.v,
                                                 (short)0, c, false, false);
}
// B fragment: lane slot = 16 contiguous bf16 (32B) in the swizzled buffer.
__device__ __forceinline__ Frag load_bfrag(const unsigned short* base,
                                           long fragIdx, int lane) {
  Frag f;
  const u32x4* p = (const u32x4*)(base + (fragIdx << 9)) + (lane << 1);
  f.q[0] = p[0];
  f.q[1] = p[1];
  return f;
}
// A fragment from a row-major LDS tile: lane holds row M=L%16,
// elements 0..7 = K[koff..koff+7], 8..15 = K[koff+16..koff+23], koff=kt*32+(L/16)*8.
__device__ __forceinline__ Frag load_afrag(const unsigned short* rowptr, int koff) {
  Frag f;
  f.q[0] = *(const u32x4*)(rowptr + koff);
  f.q[1] = *(const u32x4*)(rowptr + koff + 16);
  return f;
}

// One K-step of a 16x(32)x128 GEMM pass: preload 8 B fragments (16 clustered
// global_load_b128), then 8 back-to-back WMMAs against one A fragment.
__device__ __forceinline__ void gemm_step8(const unsigned short* rowptr, int koff,
                                           const unsigned short* wBase, long fragBase,
                                           int lane, v8f* acc) {
  Frag a = load_afrag(rowptr, koff);
  Frag bw[8];
#pragma unroll
  for (int nt = 0; nt < 8; ++nt) bw[nt] = load_bfrag(wBase, fragBase + nt, lane);
#pragma unroll
  for (int nt = 0; nt < 8; ++nt) acc[nt] = wmma_bf16(a, bw[nt], acc[nt]);
}

// ---- weight packing: f32 -> bf16, WMMA-B-fragment swizzle ----------------
// dst element (idx): e=idx&15, lane=(idx>>4)&31, frag=idx>>9;
// frag = ((agent*KT)+kt)*NT+nt ; logical (K,N) = (kt*32+(lane/16)*8+(e&7)+(e>>3)*16, nt*16+lane%16)

__global__ __launch_bounds__(256) void pack_weights(
    const float* __restrict__ src, unsigned short* __restrict__ dst,
    int Ksrc, int Kpad, int Ncols, int mode, long total)
{
  long idx = (long)blockIdx.x * 256 + threadIdx.x;
  if (idx >= total) return;
  int e    = (int)(idx & 15);
  int lane = (int)((idx >> 4) & 31);
  long f   = idx >> 9;
  int NT = Ncols >> 4, KT = Kpad >> 5;
  int nt = (int)(f % NT);
  long t2 = f / NT;
  int kt = (int)(t2 % KT);
  int agent = (int)(t2 / KT);
  int col  = nt * 16 + (lane & 15);
  int krow = kt * 32 + (lane >> 4) * 8 + (e & 7) + (e >> 3) * 16;
  float v = 0.f;
  if (krow < Ksrc) {
    long so = (mode == 0)
      ? ((long)agent * Ksrc * Ncols + (long)krow * Ncols + col)           // [A][K][N]
      : ((long)(col >> 5) * Ksrc * 32 + (long)krow * 32 + (col & 31));    // [H][K][32] -> col=h*32+d
    v = src[so];
  }
  dst[idx] = f2bf(v);
}

// ---- stage 1: encoders + K/Q/V -------------------------------------------
// block = 128 threads (4 waves); wave handles one agent x 16-row batch tile.

__global__ __launch_bounds__(128) void encode_kernel(
    const float* __restrict__ states, const float* __restrict__ actions,
    const float* __restrict__ be, const float* __restrict__ bs,
    const float* __restrict__ bv,
    const unsigned short* __restrict__ WeP, const unsigned short* __restrict__ WsP,
    const unsigned short* __restrict__ WkP, const unsigned short* __restrict__ WqP,
    const unsigned short* __restrict__ WvP,
    unsigned short* __restrict__ sEnc, unsigned short* __restrict__ Kb,
    unsigned short* __restrict__ Qb,   unsigned short* __restrict__ Vb)
{
  __shared__ unsigned short sInp[4][16][IPAD];  // inputs, later reused for K
  __shared__ unsigned short sSa [4][16][HID];   // sa_enc, later reused for V
  __shared__ unsigned short sS  [4][16][HID];   // s_enc,  later reused for Q

  const int wave = threadIdx.x >> 5, lane = threadIdx.x & 31;
  const int n = blockIdx.y;
  const long b0 = ((long)blockIdx.x * 4 + wave) * 16;
  const int row = lane & 15, half = lane >> 4;
  const int kh = half * 8;

  // warm L2 for the tile this wave touches next block-column over
  __builtin_prefetch(states + ((long)n * BATCH + b0 + 64 + row) * SDIM + half * 64, 0, 3);

  // stage [states|actions|0-pad] tile as bf16 (2 lanes per row)
  {
    const float* sp = states + ((long)n * BATCH + b0 + row) * SDIM + half * 64;
#pragma unroll
    for (int c = 0; c < 64; c += 2)
      *(unsigned*)&sInp[wave][row][half * 64 + c] = pk2(sp[c], sp[c + 1]);
    const float* ap = actions + ((long)n * BATCH + b0 + row) * ADIM + half * 8;
#pragma unroll
    for (int c = 0; c < 8; c += 2)
      *(unsigned*)&sInp[wave][row][SDIM + half * 8 + c] = pk2(ap[c], ap[c + 1]);
#pragma unroll
    for (int c = 0; c < 8; c += 2)
      *(unsigned*)&sInp[wave][row][IDIM + half * 8 + c] = 0u;
  }
  __syncthreads();

  // ---- pass 1: sa_enc = relu(inp@We + be) --------------------------------
  v8f acc[8];
#pragma unroll
  for (int nt = 0; nt < 8; ++nt) acc[nt] = vzero8();
  for (int kt = 0; kt < 5; ++kt)
    gemm_step8(&sInp[wave][row][0], kt * 32 + kh, WeP, (long)n * 40 + kt * 8, lane, acc);
#pragma unroll
  for (int nt = 0; nt < 8; ++nt) {
    const int col = nt * 16 + row;
    const float bE = be[n * HID + col];
#pragma unroll
    for (int r = 0; r < 8; ++r) {
      float v1 = acc[nt][r] + bE; v1 = v1 > 0.f ? v1 : 0.f;
      sSa[wave][kh + r][col] = f2bf(v1);
    }
  }

  // ---- pass 2: s_enc = relu(states@Ws + bs) ------------------------------
#pragma unroll
  for (int nt = 0; nt < 8; ++nt) acc[nt] = vzero8();
  for (int kt = 0; kt < 4; ++kt)
    gemm_step8(&sInp[wave][row][0], kt * 32 + kh, WsP, (long)n * 32 + kt * 8, lane, acc);
#pragma unroll
  for (int nt = 0; nt < 8; ++nt) {
    const int col = nt * 16 + row;
    const float bS = bs[n * HID + col];
#pragma unroll
    for (int r = 0; r < 8; ++r) {
      float v2 = acc[nt][r] + bS; v2 = v2 > 0.f ? v2 : 0.f;
      sS[wave][kh + r][col] = f2bf(v2);
    }
  }

  // store s_enc (bf16, coalesced)
  {
    u32x4* dst = (u32x4*)(sEnc + ((long)n * BATCH + b0 + row) * HID + half * 64);
    const u32x4* src = (const u32x4*)&sS[wave][row][half * 64];
#pragma unroll
    for (int i = 0; i < 8; ++i) dst[i] = src[i];
  }

  // ---- pass 3: K = sa_enc@Wk  (dump into sInp cols 0..127) ---------------
#pragma unroll
  for (int nt = 0; nt < 8; ++nt) acc[nt] = vzero8();
  for (int kt = 0; kt < 4; ++kt)
    gemm_step8(&sSa[wave][row][0], kt * 32 + kh, WkP, kt * 8, lane, acc);
#pragma unroll
  for (int nt = 0; nt < 8; ++nt) {
    const int col = nt * 16 + row;
#pragma unroll
    for (int r = 0; r < 8; ++r) sInp[wave][kh + r][col] = f2bf(acc[nt][r]);
  }

  // ---- pass 4: V = relu(sa_enc@Wv + bv)  (dump into sSa) -----------------
#pragma unroll
  for (int nt = 0; nt < 8; ++nt) acc[nt] = vzero8();
  for (int kt = 0; kt < 4; ++kt)
    gemm_step8(&sSa[wave][row][0], kt * 32 + kh, WvP, kt * 8, lane, acc);
#pragma unroll
  for (int nt = 0; nt < 8; ++nt) {
    const int col = nt * 16 + row;
    const float bvv = bv[(col >> 5) * AD + (col & 31)];
#pragma unroll
    for (int r = 0; r < 8; ++r) {
      float vv = acc[nt][r] + bvv; vv = vv > 0.f ? vv : 0.f;
      sSa[wave][kh + r][col] = f2bf(vv);
    }
  }

  // ---- pass 5: Q = s_enc@Wq  (dump into sS) ------------------------------
#pragma unroll
  for (int nt = 0; nt < 8; ++nt) acc[nt] = vzero8();
  for (int kt = 0; kt < 4; ++kt)
    gemm_step8(&sS[wave][row][0], kt * 32 + kh, WqP, kt * 8, lane, acc);
#pragma unroll
  for (int nt = 0; nt < 8; ++nt) {
    const int col = nt * 16 + row;
#pragma unroll
    for (int r = 0; r < 8; ++r) sS[wave][kh + r][col] = f2bf(acc[nt][r]);
  }
  __syncthreads();

  // coalesced K/V/Q stores (bf16)
  {
    const long base = ((long)n * BATCH + b0 + row) * HID + half * 64;
    const u32x4* k = (const u32x4*)&sInp[wave][row][half * 64];
    const u32x4* v = (const u32x4*)&sSa[wave][row][half * 64];
    const u32x4* q = (const u32x4*)&sS[wave][row][half * 64];
    u32x4* K4 = (u32x4*)(Kb + base);
    u32x4* V4 = (u32x4*)(Vb + base);
    u32x4* Q4 = (u32x4*)(Qb + base);
#pragma unroll
    for (int i = 0; i < 8; ++i) { K4[i] = k[i]; V4[i] = v[i]; Q4[i] = q[i]; }
  }
}

// ---- stage 2: cross-agent attention (VALU; ~1% of FLOPs) -----------------
// block = 32 threads (1 wave) handles 8 batch rows x 4 heads.

__global__ __launch_bounds__(32) void attn_kernel(
    const unsigned short* __restrict__ Kb, const unsigned short* __restrict__ Qb,
    const unsigned short* __restrict__ Vb, unsigned short* __restrict__ Ob)
{
  __shared__ unsigned short sK[N_AG][8][HID];
  __shared__ unsigned short sQ[N_AG][8][HID];
  __shared__ unsigned short sV[N_AG][8][HID];
  const long b0 = (long)blockIdx.x * 8;

  // warm L2 one tile ahead
  __builtin_prefetch(Kb + ((long)(threadIdx.x & 7) * BATCH + b0 + 8) * HID, 0, 3);

  for (int i = threadIdx.x; i < 1024; i += 32) {   // 1024 x 16B per array
    int n = i >> 7, rem = i & 127, r = rem >> 4, c4 = rem & 15;
    long g4 = ((long)n * BATCH + b0 + r) * 16 + c4;
    ((u32x4*)sK)[i] = ((const u32x4*)Kb)[g4];
    ((u32x4*)sQ)[i] = ((const u32x4*)Qb)[g4];
    ((u32x4*)sV)[i] = ((const u32x4*)Vb)[g4];
  }
  __syncthreads();

  const int h = threadIdx.x >> 3, b = threadIdx.x & 7;
  const int co = h * AD;   // 32 bf16 = 16 dwords per (agent,row) slice

  float logits[N_AG][N_AG];
#pragma unroll
  for (int i = 0; i < N_AG; ++i) {
    const unsigned* qp = (const unsigned*)&sQ[i][b][co];
    unsigned q[16];
#pragma unroll
    for (int t = 0; t < 16; ++t) q[t] = qp[t];
#pragma unroll
    for (int j = 0; j < N_AG; ++j) {
      const unsigned* kp = (const unsigned*)&sK[j][b][co];
      float acc = 0.f;
#pragma unroll
      for (int t = 0; t < 16; ++t) {
        unsigned kk = kp[t];
        acc += bflo(q[t]) * bflo(kk) + bfhi(q[t]) * bfhi(kk);
      }
      logits[i][j] = acc * 0.17677669529663687f;   // 1/sqrt(32)
    }
  }

#pragma unroll
  for (int i = 0; i < N_AG; ++i) {
    float mx = -3.4e38f;
#pragma unroll
    for (int j = 0; j < N_AG; ++j)
      if (j != i && logits[i][j] > mx) mx = logits[i][j];
    float w[N_AG], s = 0.f;
#pragma unroll
    for (int j = 0; j < N_AG; ++j) {
      float e = (j == i) ? 0.f : __expf(logits[i][j] - mx);
      w[j] = e; s += e;
    }
    const float inv = 1.f / s;
    float acc[32];
#pragma unroll
    for (int t = 0; t < 32; ++t) acc[t] = 0.f;
#pragma unroll
    for (int j = 0; j < N_AG; ++j) {
      if (j == i) continue;
      const float a = w[j] * inv;
      const unsigned* vp = (const unsigned*)&sV[j][b][co];
#pragma unroll
      for (int t = 0; t < 16; ++t) {
        unsigned vv = vp[t];
        acc[2 * t]     += a * bflo(vv);
        acc[2 * t + 1] += a * bfhi(vv);
      }
    }
    unsigned* op = (unsigned*)(Ob + ((long)i * BATCH + b0 + b) * HID + co);
#pragma unroll
    for (int t = 0; t < 16; ++t) op[t] = pk2(acc[2 * t], acc[2 * t + 1]);
  }
}

// ---- stage 3: critic MLP + argmax gather ---------------------------------

__global__ __launch_bounds__(128) void critic_kernel(
    const unsigned short* __restrict__ sEnc, const unsigned short* __restrict__ Ob,
    const unsigned short* __restrict__ Wc1P, const unsigned short* __restrict__ Wc2P,
    const float* __restrict__ bc1, const float* __restrict__ bc2,
    const float* __restrict__ actions, float* __restrict__ out)
{
  __shared__ unsigned short sIn[4][16][2 * HID];
  __shared__ unsigned short sH [4][16][HID];
  __shared__ float          sQv[4][16][16];

  const int wave = threadIdx.x >> 5, lane = threadIdx.x & 31;
  const int n = blockIdx.y;
  const long b0 = ((long)blockIdx.x * 4 + wave) * 16;
  const int row = lane & 15, half = lane >> 4;
  const int kh = half * 8;

  __builtin_prefetch(sEnc + ((long)n * BATCH + b0 + 64 + row) * HID + half * 64, 0, 3);
  __builtin_prefetch(Ob   + ((long)n * BATCH + b0 + 64 + row) * HID + half * 64, 0, 3);

  // stage critic_in = [s_enc | other]
  {
    const u32x4* se = (const u32x4*)(sEnc + ((long)n * BATCH + b0 + row) * HID + half * 64);
    u32x4* d1 = (u32x4*)&sIn[wave][row][half * 64];
#pragma unroll
    for (int i = 0; i < 8; ++i) d1[i] = se[i];
    const u32x4* ob = (const u32x4*)(Ob + ((long)n * BATCH + b0 + row) * HID + half * 64);
    u32x4* d2 = (u32x4*)&sIn[wave][row][HID + half * 64];
#pragma unroll
    for (int i = 0; i < 8; ++i) d2[i] = ob[i];
  }
  __syncthreads();

  // h1 = relu(critic_in @ Wc1 + bc1)
  v8f acc[8];
#pragma unroll
  for (int nt = 0; nt < 8; ++nt) acc[nt] = vzero8();
  for (int kt = 0; kt < 8; ++kt)
    gemm_step8(&sIn[wave][row][0], kt * 32 + kh, Wc1P, (long)n * 64 + kt * 8, lane, acc);
#pragma unroll
  for (int nt = 0; nt < 8; ++nt) {
    const int col = nt * 16 + row;
    const float bb = bc1[n * HID + col];
#pragma unroll
    for (int r = 0; r < 8; ++r) {
      float v = acc[nt][r] + bb; v = v > 0.f ? v : 0.f;
      sH[wave][kh + r][col] = f2bf(v);
    }
  }
  __syncthreads();

  // all_q = h1 @ Wc2 + bc2  (single 16-wide N tile; preload the 4 B frags)
  v8f aq = vzero8();
  {
    Frag bw[4];
#pragma unroll
    for (int kt = 0; kt < 4; ++kt) bw[kt] = load_bfrag(Wc2P, (long)n * 4 + kt, lane);
#pragma unroll
    for (int kt = 0; kt < 4; ++kt) {
      Frag a = load_afrag(&sH[wave][row][0], kt * 32 + kh);
      aq = wmma_bf16(a, bw[kt], aq);
    }
  }
  {
    const float bb = bc2[n * ADIM + row];
#pragma unroll
    for (int r = 0; r < 8; ++r) sQv[wave][kh + r][row] = aq[r] + bb;
  }
  __syncthreads();

  // argmax over actions, gather q
  if (lane < 16) {
    const long b = b0 + lane;
    const float* ap = actions + ((long)n * BATCH + b) * ADIM;
    int best = 0; float bmax = ap[0];
#pragma unroll
    for (int c = 1; c < ADIM; ++c) {
      float v = ap[c];
      if (v > bmax) { bmax = v; best = c; }
    }
    out[(long)n * BATCH + b] = sQv[wave][lane][best];
  }
}

// ---- launcher ------------------------------------------------------------

extern "C" void kernel_launch(void* const* d_in, const int* in_sizes, int n_in,
                              void* d_out, int out_size, void* d_ws, size_t ws_size,
                              hipStream_t stream)
{
  const float* states  = (const float*)d_in[0];
  const float* actions = (const float*)d_in[1];
  const float* We  = (const float*)d_in[2];
  const float* be  = (const float*)d_in[3];
  const float* Ws  = (const float*)d_in[4];
  const float* bs  = (const float*)d_in[5];
  const float* Wk  = (const float*)d_in[6];
  const float* Wq  = (const float*)d_in[7];
  const float* Wv  = (const float*)d_in[8];
  const float* bv  = (const float*)d_in[9];
  const float* Wc1 = (const float*)d_in[10];
  const float* bc1 = (const float*)d_in[11];
  const float* Wc2 = (const float*)d_in[12];
  const float* bc2 = (const float*)d_in[13];
  float* out = (float*)d_out;

  unsigned short* ws = (unsigned short*)d_ws;
  size_t o = 0;
  unsigned short* WeP  = ws + o; o += (size_t)8 * 5 * 8 * 512;   // [N][160][128]
  unsigned short* WsP  = ws + o; o += (size_t)8 * 4 * 8 * 512;   // [N][128][128]
  unsigned short* WkP  = ws + o; o += (size_t)4 * 8 * 512;       // [128][128] head-packed
  unsigned short* WqP  = ws + o; o += (size_t)4 * 8 * 512;
  unsigned short* WvP  = ws + o; o += (size_t)4 * 8 * 512;
  unsigned short* Wc1P = ws + o; o += (size_t)8 * 8 * 8 * 512;   // [N][256][128]
  unsigned short* Wc2P = ws + o; o += (size_t)8 * 4 * 1 * 512;   // [N][128][16]
  const size_t actSz = (size_t)N_AG * BATCH * HID;               // 33.5M bf16 each
  unsigned short* sEnc = ws + o; o += actSz;
  unsigned short* Kb   = ws + o; o += actSz;
  unsigned short* Qb   = ws + o; o += actSz;
  unsigned short* Vb   = ws + o; o += actSz;
  unsigned short* Ob   = ws + o; o += actSz;

  auto pk = [&](const float* src, unsigned short* dst, int Ksrc, int Kpad,
                int Ncols, int mode, int nAg) {
    long total = (long)nAg * (Kpad / 32) * (Ncols / 16) * 512;
    pack_weights<<<dim3((unsigned)((total + 255) / 256)), 256, 0, stream>>>(
        src, dst, Ksrc, Kpad, Ncols, mode, total);
  };
  pk(We,  WeP,  144, 160, 128, 0, 8);
  pk(Ws,  WsP,  128, 128, 128, 0, 8);
  pk(Wk,  WkP,  128, 128, 128, 1, 1);
  pk(Wq,  WqP,  128, 128, 128, 1, 1);
  pk(Wv,  WvP,  128, 128, 128, 1, 1);
  pk(Wc1, Wc1P, 256, 256, 128, 0, 8);
  pk(Wc2, Wc2P, 128, 128, 16,  0, 8);

  encode_kernel<<<dim3(BATCH / 64, N_AG), 128, 0, stream>>>(
      states, actions, be, bs, bv, WeP, WsP, WkP, WqP, WvP, sEnc, Kb, Qb, Vb);
  attn_kernel<<<dim3(BATCH / 8), 32, 0, stream>>>(Kb, Qb, Vb, Ob);
  critic_kernel<<<dim3(BATCH / 64, N_AG), 128, 0, stream>>>(
      sEnc, Ob, Wc1P, Wc2P, bc1, bc2, actions, out);

  (void)in_sizes; (void)n_in; (void)out_size; (void)ws_size;
}
